// YatPerformerLaplaceCausalAttention_56367150793420
// MI455X (gfx1250) — matrix operand order repaired and checked by reading, not compiled, not run on hardware
//
#include <hip/hip_runtime.h>
#include <math.h>
#include <stdint.h>

// ---------------------------------------------------------------------------
// CDNA5 (gfx1250) WMMA types & helpers
// ---------------------------------------------------------------------------
typedef __attribute__((ext_vector_type(16))) __bf16 bf16x16;
typedef __attribute__((ext_vector_type(8)))  float  f32x8;
typedef __attribute__((ext_vector_type(4)))  unsigned int uint32x4;
typedef __attribute__((ext_vector_type(8)))  int int32x8;
typedef __attribute__((ext_vector_type(4)))  int int32x4;

#define DEVINL __device__ __forceinline__

#if __has_builtin(__builtin_amdgcn_tensor_load_to_lds) && \
    __has_builtin(__builtin_amdgcn_s_wait_tensorcnt)
#define HAVE_TDM 1
#else
#define HAVE_TDM 0
#endif

DEVINL unsigned short f2bf(float f) {
  unsigned u = __float_as_uint(f);
  unsigned r = (u + 0x7FFFu + ((u >> 16) & 1u)) >> 16;  // round-to-nearest-even
  return (unsigned short)r;
}
DEVINL float bf2f(unsigned short h) { return __uint_as_float(((unsigned)h) << 16); }
DEVINL __bf16 bfbits(unsigned short h) { return __builtin_bit_cast(__bf16, h); }

DEVINL f32x8 wmma_bf16(bf16x16 a, bf16x16 b, f32x8 c) {
  return __builtin_amdgcn_wmma_f32_16x16x32_bf16(false, a, false, b, (short)0, c,
                                                 false, false);
}

// Fragment-major ("swizzled") offsets within one subtile, element units.
// A-subtile = 16(M) x 32(K): lane = (m&15)+16*((k>>3)&1); p = 2j+(k&1),
//   j = ((k>>1)&3) + 4*((k>>4)&1)
DEVINL int a_off(int m, int k) {
  int h = (k >> 3) & 1;
  int j = ((k >> 1) & 3) + 4 * ((k >> 4) & 1);
  return (((m & 15) + 16 * h) << 4) + 2 * j + (k & 1);
}
// B-subtile = 32(K) x 16(N): lane = (n&15)+16*((k>>3)&1); p = 2j+(k&1),
//   j = ((k>>1)&3) + 4*((k>>4)&1)
DEVINL int b_off(int k, int n) {
  int h = (k >> 3) & 1;
  int j = ((k >> 1) & 3) + 4 * ((k >> 4) & 1);
  return (((n & 15) + 16 * h) << 4) + 2 * j + (k & 1);
}

// Load a fragment: lane's 16 contiguous bf16 (32 bytes) at subtile base.
DEVINL bf16x16 frag_ld(const unsigned short* sub) {
  unsigned l = threadIdx.x & 31u;
  return *(const bf16x16*)(sub + l * 16);
}
// Constant B fragment: column 0 of the tile = 1.0, all other cols 0.
DEVINL bf16x16 frag_ones_col0() {
  unsigned l = threadIdx.x & 31u;
  unsigned short v = ((l & 15u) == 0) ? (unsigned short)0x3F80 : (unsigned short)0;
  bf16x16 r;
#pragma unroll
  for (int i = 0; i < 16; ++i) r[i] = bfbits(v);
  return r;
}

#if HAVE_TDM
// Issue a TDM 1-row copy of `nelem` bf16 elements (global -> LDS).
// D# per CDNA5 ISA ch.8: group0 = {count|flags, lds_addr, global_addr, type=2},
// group1 = {mask/data_size, tensor_dim0/1, tile_dim0/1/2, strides}.
DEVINL void tdm_load_bf16(unsigned lds_off, const void* gsrc, unsigned nelem) {
  unsigned long long ga = (unsigned long long)(uintptr_t)gsrc;
  uint32x4 g0;
  g0[0] = 1u;                                             // count=1 (valid), user mode
  g0[1] = lds_off;                                        // lds_addr [63:32]
  g0[2] = (unsigned)(ga & 0xffffffffu);                   // global_addr [95:64]
  g0[3] = (unsigned)((ga >> 32) & 0x01ffffffu) | (2u << 30);  // addr[56:32], type=2
  int32x8 g1;
  g1[0] = (int)(1u << 16);                                // data_size=1 (2 bytes)
  g1[1] = (int)((nelem & 0xffffu) << 16);                 // tensor_dim0[15:0]
  g1[2] = (int)(((nelem >> 16) & 0xffffu) | (1u << 16));  // tensor_dim0[31:16], dim1=1
  g1[3] = (int)((nelem & 0xffffu) << 16);                 // tile_dim0
  g1[4] = 1;                                              // tile_dim1=1, tile_dim2=0
  g1[5] = (int)nelem;                                     // tensor_dim0_stride[31:0]
  g1[6] = 0;
  g1[7] = 0;
  int32x4 z = {0, 0, 0, 0};
#if defined(__clang_major__) && (__clang_major__ >= 23)
  int32x8 z8 = {0, 0, 0, 0, 0, 0, 0, 0};
  __builtin_amdgcn_tensor_load_to_lds(g0, g1, z, z, z8, 0);
#else
  __builtin_amdgcn_tensor_load_to_lds(g0, g1, z, z, 0);
#endif
}
#endif

// ---------------------------------------------------------------------------
// 1) Converts: fp32 -> bf16 directly into fragment-major global layouts.
// ---------------------------------------------------------------------------
__global__ __launch_bounds__(256) void k_cvt_a(const float* __restrict__ in,
                                               unsigned short* __restrict__ out,
                                               int n) {
  int i = blockIdx.x * 256 + threadIdx.x;
  if (i >= n) return;
  int r = i >> 10, k = i & 1023;
  int o = (((r >> 4) * 32 + (k >> 5)) << 9) + a_off(r, k & 31);
  out[o] = f2bf(in[i]);
}
__global__ __launch_bounds__(256) void k_cvt_b(const float* __restrict__ in,
                                               unsigned short* __restrict__ out,
                                               int N, int n) {
  int i = blockIdx.x * 256 + threadIdx.x;
  if (i >= n) return;
  int k = i / N, c = i - k * N;
  int o = ((((c >> 6) * 32 + (k >> 5)) * 4 + ((c >> 4) & 3)) << 9) + b_off(k & 31, c);
  out[o] = f2bf(in[i]);
}
// omega (R=2,H=16,D=64,M=32) -> per-head B-swizzled bf16 Om[d][f=r*32+m],
// scaled by sqrt(2*max(s_r,0)).
__global__ __launch_bounds__(256) void k_cvt_om(const float* __restrict__ omega,
                                                const float* __restrict__ nodes,
                                                unsigned short* __restrict__ om_b) {
  int i = blockIdx.x * 256 + threadIdx.x;  // 65536 total
  if (i >= 65536) return;
  int r = i >> 15;
  int rem = i & 32767;
  int h = rem >> 11;
  int rem2 = rem & 2047;
  int d = rem2 >> 5, m = rem2 & 31;
  int f = r * 32 + m;
  float scale = sqrtf(2.f * fmaxf(nodes[r], 0.f));
  int o = (h << 12) + (((d >> 5) * 4 + (f >> 4)) << 9) + b_off(d & 31, f);
  om_b[o] = f2bf(omega[i] * scale);
}

// ---------------------------------------------------------------------------
// 2) QKV GEMM (4096x1024)x(1024x3072)+bias. No LDS; 32B-fragment global loads;
//    K-loop software-pipelined.
// ---------------------------------------------------------------------------
__global__ __launch_bounds__(128) void k_qkv_gemm(
    const unsigned short* __restrict__ xa, const unsigned short* __restrict__ wb,
    const float* __restrict__ bias, unsigned short* __restrict__ q_rm,
    unsigned short* __restrict__ k_rm, unsigned short* __restrict__ v_sw) {
  const int ncb = blockIdx.x;
  const int row0 = blockIdx.y * 64;
  const int wave = threadIdx.x >> 5;
  const unsigned l = threadIdx.x & 31u;
  const int lh = l >> 4, ln = l & 15;
  const int mt = (row0 >> 4) + wave;
  const unsigned short* abase = xa + (((size_t)mt * 32) << 9);
  const unsigned short* bbase = wb + (((size_t)ncb * 128) << 9);
  f32x8 acc[4] = {{}, {}, {}, {}};
  bf16x16 a = frag_ld(abase);
  bf16x16 b0 = frag_ld(bbase), b1 = frag_ld(bbase + 512);
  bf16x16 b2 = frag_ld(bbase + 1024), b3 = frag_ld(bbase + 1536);
  for (int kt = 0; kt < 31; ++kt) {
    const unsigned short* ap = abase + ((kt + 1) << 9);
    const unsigned short* bp = bbase + (((size_t)(kt + 1) * 4) << 9);
    bf16x16 an = frag_ld(ap);
    bf16x16 bn0 = frag_ld(bp), bn1 = frag_ld(bp + 512);
    bf16x16 bn2 = frag_ld(bp + 1024), bn3 = frag_ld(bp + 1536);
    acc[0] = wmma_bf16(a, b0, acc[0]);
    acc[1] = wmma_bf16(a, b1, acc[1]);
    acc[2] = wmma_bf16(a, b2, acc[2]);
    acc[3] = wmma_bf16(a, b3, acc[3]);
    a = an; b0 = bn0; b1 = bn1; b2 = bn2; b3 = bn3;
  }
  acc[0] = wmma_bf16(a, b0, acc[0]);
  acc[1] = wmma_bf16(a, b1, acc[1]);
  acc[2] = wmma_bf16(a, b2, acc[2]);
  acc[3] = wmma_bf16(a, b3, acc[3]);
#pragma unroll
  for (int cb = 0; cb < 4; ++cb) {
#pragma unroll
    for (int v = 0; v < 8; ++v) {
      int r = row0 + wave * 16 + v + 8 * lh;
      int cc = ncb * 64 + cb * 16 + ln;
      float val = acc[cb][v] + bias[cc];
      int which = cc >> 10, e = cc & 1023;
      int head = e >> 6, d = e & 63;
      int b_ = r >> 11, t = r & 2047;
      unsigned short bv = f2bf(val);
      int bh = b_ * 16 + head;
      if (which == 2) {
        int tl = t & 63;
        size_t o = (((size_t)(bh * 32 + (t >> 6)) * 8 + (tl >> 5) * 4 + (d >> 4)) << 9) +
                   b_off(tl & 31, d);
        v_sw[o] = bv;
      } else {
        size_t o = (((size_t)bh * 2048) + t) * 64 + d;
        if (which == 0) q_rm[o] = bv; else k_rm[o] = bv;
      }
    }
  }
}

// ---------------------------------------------------------------------------
// 3) Laplace features via WMMA.
// ---------------------------------------------------------------------------
__global__ __launch_bounds__(128) void k_feat(
    const unsigned short* __restrict__ q_rm, const unsigned short* __restrict__ k_rm,
    const unsigned short* __restrict__ om_b, const float* __restrict__ nodes,
    const float* __restrict__ weights, unsigned short* __restrict__ qf_a,
    unsigned short* __restrict__ kf_bT, unsigned short* __restrict__ kf_aT) {
  __shared__ float nrmInv[64];
  __shared__ __align__(32) unsigned short Xn[4096];  // 64x64 A-swz
  const int c = blockIdx.x;
  const int bh = blockIdx.y;
  const int h = bh & 15;
  const int isK = blockIdx.z;
  const unsigned short* src = isK ? k_rm : q_rm;
  const int tid = threadIdx.x;
  const int wave = tid >> 5;
  const unsigned l = tid & 31u;
  const int lh = l >> 4, ln = l & 15;
  const int m0 = wave * 16;
  const size_t rbase = (((size_t)bh * 2048) + (size_t)c * 64) << 6;
  const float s0 = nodes[0], s1 = nodes[1];
  const float c0 = sqrtf(fmaxf(weights[0], 0.f)) * 0.17677669529663687f;
  const float c1 = sqrtf(fmaxf(weights[1], 0.f)) * 0.17677669529663687f;

  if (tid < 64) {
    float nrm = 0.f;
#pragma unroll
    for (int d = 0; d < 64; ++d) {
      float xv = bf2f(src[rbase + tid * 64 + d]);
      nrm += xv * xv;
    }
    nrmInv[tid] = 1.f / fmaxf(sqrtf(nrm), 1e-12f);
  }
  __syncthreads();
  for (int i = tid; i < 4096; i += 128) {
    int r = i >> 6, d = i & 63;
    float xv = bf2f(src[rbase + i]) * nrmInv[r];
    Xn[(((r >> 4) * 2 + (d >> 5)) << 9) + a_off(r, d & 31)] = f2bf(xv);
  }
  __syncthreads();

  const size_t cbase = ((size_t)(bh * 32 + c)) << 12;
  const unsigned short* om_h = om_b + (h << 12);
#pragma unroll
  for (int cb = 0; cb < 4; ++cb) {
    f32x8 acc = {};
#pragma unroll
    for (int kt = 0; kt < 2; ++kt) {
      bf16x16 a = frag_ld(Xn + ((wave * 2 + kt) << 9));
      bf16x16 b = frag_ld(om_h + ((kt * 4 + cb) << 9));
      acc = wmma_bf16(a, b, acc);
    }
#pragma unroll
    for (int v = 0; v < 8; ++v) {
      int rloc = m0 + v + 8 * lh;
      int f = cb * 16 + ln;
      float sr = (f >= 32) ? s1 : s0;
      float cs = (f >= 32) ? c1 : c0;
      float arg = fminf(fmaxf(acc[v] - sr, -20.f), 20.f);
      unsigned short bv = f2bf(__expf(arg) * cs);
      if (!isK) {
        qf_a[cbase + (((rloc >> 4) * 2 + (f >> 5)) << 9) + a_off(rloc, f & 31)] = bv;
      } else {
        kf_bT[cbase + (((f >> 5) * 4 + (rloc >> 4)) << 9) + b_off(f & 31, rloc)] = bv;
        kf_aT[cbase + (((f >> 4) * 2 + (rloc >> 5)) << 9) + a_off(f, rloc & 31)] = bv;
      }
    }
  }
}

// ---------------------------------------------------------------------------
// 4) Chunked causal linear-attention scan. One WG per (b,h), 4 waves.
//    V̂ = [V | 1]; KV' = [KV | k_state] (64x80 fp32 + bf16 shadow in LDS).
//    V tiles are prefetched one chunk ahead with the Tensor Data Mover
//    (tensor_load_to_lds + s_wait_tensorcnt), double-buffered in LDS.
// ---------------------------------------------------------------------------
__global__ __launch_bounds__(128) void k_scan(
    const unsigned short* __restrict__ qf_a, const unsigned short* __restrict__ kf_bT,
    const unsigned short* __restrict__ kf_aT, const unsigned short* __restrict__ v_sw,
    unsigned short* __restrict__ attn_a) {
  __shared__ __align__(32) float KVf[10 * 512];          // 64x80 fp32
  __shared__ __align__(32) unsigned short KVb[10 * 512]; // bf16 shadow
  __shared__ __align__(32) unsigned short Ss[8 * 512];   // tril(S), A-swz
  __shared__ __align__(32) unsigned short Vt[2][4096];   // TDM-staged V tiles
  __shared__ float den[64];
  const int bh = blockIdx.x;
  const int b = bh >> 4, h = bh & 15;
  const int tid = threadIdx.x;
  const int wave = tid >> 5;
  const unsigned l = tid & 31u;
  const int lh = l >> 4, ln = l & 15;
  const int m0 = wave * 16;

  for (int i = tid; i < 10 * 512; i += 128) {
    KVf[i] = 0.f;
    KVb[i] = 0;
  }
#if HAVE_TDM
  if (wave == 0) {
    tdm_load_bf16((unsigned)(uintptr_t)(&Vt[0][0]),
                  v_sw + (((size_t)(bh * 32)) << 12), 4096);
  }
#endif
  __syncthreads();

  for (int c = 0; c < 32; ++c) {
    const size_t cbase = ((size_t)(bh * 32 + c)) << 12;

#if HAVE_TDM
    if (wave == 0) {
      if (c + 1 < 32) {
        tdm_load_bf16((unsigned)(uintptr_t)(&Vt[(c + 1) & 1][0]),
                      v_sw + (((size_t)(bh * 32 + c + 1)) << 12), 4096);
        __builtin_amdgcn_s_wait_tensorcnt(1);  // chunk c's tile has landed
      } else {
        __builtin_amdgcn_s_wait_tensorcnt(0);
      }
    }
    __syncthreads();
    const unsigned short* vtile = &Vt[c & 1][0];
#else
    const unsigned short* vtile = v_sw + cbase;
#endif

    // ---- S = tril(Qf Kf^T) ----
#pragma unroll
    for (int cb = 0; cb < 4; ++cb) {
      f32x8 acc = {};
#pragma unroll
      for (int kt = 0; kt < 2; ++kt) {
        bf16x16 a = frag_ld(qf_a + cbase + ((wave * 2 + kt) << 9));
        bf16x16 bm = frag_ld(kf_bT + cbase + ((kt * 4 + cb) << 9));
        acc = wmma_bf16(a, bm, acc);
      }
#pragma unroll
      for (int v = 0; v < 8; ++v) {
        int r = m0 + v + 8 * lh;
        int cc = cb * 16 + ln;
        Ss[((wave * 2 + (cc >> 5)) << 9) + a_off(r, cc & 31)] =
            f2bf((cc <= r) ? acc[v] : 0.f);
      }
    }
    __syncthreads();

    // ---- C' = Qf·KV' + S·V̂ over 5 col tiles; col 64 = den ----
    f32x8 cacc[5];
#pragma unroll
    for (int cb = 0; cb < 5; ++cb) {
      f32x8 acc = {};
#pragma unroll
      for (int kt = 0; kt < 2; ++kt) {
        bf16x16 aq = frag_ld(qf_a + cbase + ((wave * 2 + kt) << 9));
        bf16x16 bkv = frag_ld(KVb + ((kt * 5 + cb) << 9));
        acc = wmma_bf16(aq, bkv, acc);
        bf16x16 ap = frag_ld(Ss + ((wave * 2 + kt) << 9));
        bf16x16 bv = (cb < 4) ? frag_ld(vtile + ((kt * 4 + cb) << 9))
                              : frag_ones_col0();
        acc = wmma_bf16(ap, bv, acc);
      }
      cacc[cb] = acc;
    }
    if (ln == 0) {
#pragma unroll
      for (int v = 0; v < 8; ++v) den[m0 + v + 8 * lh] = fmaxf(cacc[4][v], 1e-6f);
    }
    __syncthreads();  // den ready; all KVb reads complete

    // ---- out = ctx/den, written in A-swizzle ----
#pragma unroll
    for (int cb = 0; cb < 4; ++cb) {
#pragma unroll
      for (int v = 0; v < 8; ++v) {
        int r = m0 + v + 8 * lh;
        float o = cacc[cb][v] / den[r];
        int rr = b * 2048 + c * 64 + r;
        int cg = h * 64 + cb * 16 + ln;
        attn_a[(((size_t)(rr >> 4) * 32 + (cg >> 5)) << 9) + a_off(rr, cg & 31)] =
            f2bf(o);
      }
    }

    // ---- KV' += Kf^T·V̂ ----
#pragma unroll
    for (int cb = 0; cb < 5; ++cb) {
      f32x8 acc;
#pragma unroll
      for (int v = 0; v < 8; ++v) {
        int f = m0 + v + 8 * lh;
        acc[v] = KVf[(((f >> 5) * 5 + cb) << 9) + b_off(f & 31, ln)];
      }
#pragma unroll
      for (int kt = 0; kt < 2; ++kt) {
        bf16x16 ak = frag_ld(kf_aT + cbase + ((wave * 2 + kt) << 9));
        bf16x16 bv = (cb < 4) ? frag_ld(vtile + ((kt * 4 + cb) << 9))
                              : frag_ones_col0();
        acc = wmma_bf16(ak, bv, acc);
      }
#pragma unroll
      for (int v = 0; v < 8; ++v) {
        int f = m0 + v + 8 * lh;
        int idx = (((f >> 5) * 5 + cb) << 9) + b_off(f & 31, ln);
        KVf[idx] = acc[v];
        KVb[idx] = f2bf(acc[v]);
      }
    }
    __syncthreads();
  }
}

// ---------------------------------------------------------------------------
// 5) Output GEMM: attn(A-swz) x Wout(B-swz) + bout -> fp32 d_out. Pipelined.
// ---------------------------------------------------------------------------
__global__ __launch_bounds__(128) void k_out_gemm(
    const unsigned short* __restrict__ aa, const unsigned short* __restrict__ wb,
    const float* __restrict__ bias, float* __restrict__ out) {
  const int ncb = blockIdx.x;
  const int row0 = blockIdx.y * 64;
  const int wave = threadIdx.x >> 5;
  const unsigned l = threadIdx.x & 31u;
  const int lh = l >> 4, ln = l & 15;
  const int mt = (row0 >> 4) + wave;
  const unsigned short* abase = aa + (((size_t)mt * 32) << 9);
  const unsigned short* bbase = wb + (((size_t)ncb * 128) << 9);
  f32x8 acc[4] = {{}, {}, {}, {}};
  bf16x16 a = frag_ld(abase);
  bf16x16 b0 = frag_ld(bbase), b1 = frag_ld(bbase + 512);
  bf16x16 b2 = frag_ld(bbase + 1024), b3 = frag_ld(bbase + 1536);
  for (int kt = 0; kt < 31; ++kt) {
    const unsigned short* ap = abase + ((kt + 1) << 9);
    const unsigned short* bp = bbase + (((size_t)(kt + 1) * 4) << 9);
    bf16x16 an = frag_ld(ap);
    bf16x16 bn0 = frag_ld(bp), bn1 = frag_ld(bp + 512);
    bf16x16 bn2 = frag_ld(bp + 1024), bn3 = frag_ld(bp + 1536);
    acc[0] = wmma_bf16(a, b0, acc[0]);
    acc[1] = wmma_bf16(a, b1, acc[1]);
    acc[2] = wmma_bf16(a, b2, acc[2]);
    acc[3] = wmma_bf16(a, b3, acc[3]);
    a = an; b0 = bn0; b1 = bn1; b2 = bn2; b3 = bn3;
  }
  acc[0] = wmma_bf16(a, b0, acc[0]);
  acc[1] = wmma_bf16(a, b1, acc[1]);
  acc[2] = wmma_bf16(a, b2, acc[2]);
  acc[3] = wmma_bf16(a, b3, acc[3]);
#pragma unroll
  for (int cb = 0; cb < 4; ++cb) {
#pragma unroll
    for (int v = 0; v < 8; ++v) {
      int r = row0 + wave * 16 + v + 8 * lh;
      int cc = ncb * 64 + cb * 16 + ln;
      out[(size_t)r * 1024 + cc] = acc[cb][v] + bias[cc];
    }
  }
}

// ---------------------------------------------------------------------------
// Host-side launch
// ---------------------------------------------------------------------------
extern "C" void kernel_launch(void* const* d_in, const int* in_sizes, int n_in,
                              void* d_out, int out_size, void* d_ws, size_t ws_size,
                              hipStream_t stream) {
  (void)in_sizes; (void)n_in; (void)out_size; (void)ws_size;
  const float* x     = (const float*)d_in[0];
  const float* Wqkv  = (const float*)d_in[1];
  const float* bqkv  = (const float*)d_in[2];
  const float* Wout  = (const float*)d_in[3];
  const float* bout  = (const float*)d_in[4];
  const float* omega = (const float*)d_in[5];
  const float* qn    = (const float*)d_in[6];
  const float* qw    = (const float*)d_in[7];
  float* out = (float*)d_out;

  char* ws = (char*)d_ws;
  size_t off = 0;
  auto take = [&](size_t nbytes) -> char* {
    char* p = ws + off;
    off += (nbytes + 255) & ~(size_t)255;
    return p;
  };
  const size_t NX  = (size_t)2 * 2048 * 1024;
  const size_t NWQ = (size_t)1024 * 3072;
  const size_t NWO = (size_t)1024 * 1024;
  const size_t NOM = (size_t)65536;
  unsigned short* x_a    = (unsigned short*)take(NX * 2);
  unsigned short* wq_b   = (unsigned short*)take(NWQ * 2);
  unsigned short* wo_b   = (unsigned short*)take(NWO * 2);
  unsigned short* om_b   = (unsigned short*)take(NOM * 2);
  unsigned short* q_rm   = (unsigned short*)take(NX * 2);
  unsigned short* k_rm   = (unsigned short*)take(NX * 2);
  unsigned short* v_sw   = (unsigned short*)take(NX * 2);
  unsigned short* qf_a   = (unsigned short*)take(NX * 2);
  unsigned short* kf_bT  = (unsigned short*)take(NX * 2);
  unsigned short* kf_aT  = (unsigned short*)take(NX * 2);
  unsigned short* attn_a = (unsigned short*)take(NX * 2);

  k_cvt_a<<<dim3((unsigned)((NX + 255) / 256)), 256, 0, stream>>>(x, x_a, (int)NX);
  k_cvt_b<<<dim3((unsigned)((NWQ + 255) / 256)), 256, 0, stream>>>(Wqkv, wq_b, 3072,
                                                                   (int)NWQ);
  k_cvt_b<<<dim3((unsigned)((NWO + 255) / 256)), 256, 0, stream>>>(Wout, wo_b, 1024,
                                                                   (int)NWO);
  k_cvt_om<<<dim3(256), 256, 0, stream>>>(omega, qn, om_b);

  k_qkv_gemm<<<dim3(48, 64), 128, 0, stream>>>(x_a, wq_b, bqkv, q_rm, k_rm, v_sw);

  k_feat<<<dim3(32, 32, 2), 128, 0, stream>>>(q_rm, k_rm, om_b, qn, qw, qf_a, kf_bT,
                                              kf_aT);

  k_scan<<<dim3(32), 128, 0, stream>>>(qf_a, kf_bT, kf_aT, v_sw, attn_a);

  k_out_gemm<<<dim3(16, 64), 128, 0, stream>>>(attn_a, wo_b, bout, out);
}
